// BidirectionalCrossModalAttention_55963423866987
// MI455X (gfx1250) — matrix operand order repaired and checked by reading, not compiled, not run on hardware
//
#include <hip/hip_runtime.h>

typedef __attribute__((ext_vector_type(16))) _Float16 v16h;
typedef __attribute__((ext_vector_type(8)))  _Float16 v8h;
typedef __attribute__((ext_vector_type(4)))  _Float16 v4h;
typedef __attribute__((ext_vector_type(8)))  float    v8f;

constexpr int   kDim   = 1024;
constexpr int   kBatch = 32768;
constexpr float kEps   = 1e-5f;

constexpr int BM = 128, BN = 64, BK = 64;
constexpr int LDA_S = BK + 16;  // 80 halves = 160B row stride (16B aligned, conflict-padded)
constexpr int LDW_S = BK + 16;

// Async global -> LDS copy of 16 bytes (CDNA5 GLOBAL_LOAD_ASYNC_TO_LDS_B128, ASYNCcnt).
// VDST = VGPR holding LDS byte address (low 32 bits of flat shared pointer), VADDR = 64b address.
__device__ __forceinline__ void async_copy_b128(const void* gsrc, void* ldst) {
  unsigned lds_addr = (unsigned)(size_t)ldst;
  asm volatile("global_load_async_to_lds_b128 %0, %1, off" :: "v"(lds_addr), "v"(gsrc) : "memory");
}

// ---- WMMA fragment loads per CDNA5 ISA 7.12.2 layouts (wave32) ----
__device__ __forceinline__ v16h load_a_frag(const _Float16* s, int rbase, int kk, int l15, int hi) {
  // A 16x32 f16: lane row = l&15; lanes<16 hold K {0..7,16..23}, lanes>=16 hold K {8..15,24..31}
  const _Float16* p = s + (rbase + l15) * LDA_S + kk + hi * 8;
  v8h lo = *(const v8h*)p;
  v8h hh = *(const v8h*)(p + 16);
  return __builtin_shufflevector(lo, hh, 0,1,2,3,4,5,6,7,8,9,10,11,12,13,14,15);
}
__device__ __forceinline__ v16h load_b_frag(const _Float16* s, int cbase, int kk, int l15, int hi) {
  // B 32x16 f16: lane col = l&15; lanes<16 hold K {0..7,16..23}, lanes>=16 hold K {8..15,24..31}
  const _Float16* p = s + (cbase + l15) * LDW_S + kk + hi * 8;
  v8h lo = *(const v8h*)p;
  v8h hh = *(const v8h*)(p + 16);
  return __builtin_shufflevector(lo, hh, 0,1,2,3,4,5,6,7,8,9,10,11,12,13,14,15);
}

// C[m,n] = sum_k A[m,k] * W[n,k] (+ bias[n]); A:[gridM*BM,K] lda, W:[gridN*BN,K] ldw row-major.
// Double-buffered LDS; next K-tile fetched via async-to-LDS while current tile runs on WMMA.
template <typename OutT>
__global__ __launch_bounds__(256) void gemm_wmma(const _Float16* __restrict__ A, int lda,
                                                 const _Float16* __restrict__ W, int ldw,
                                                 OutT* __restrict__ C, int ldc,
                                                 const float* __restrict__ bias, int K) {
  __shared__ __align__(16) _Float16 sA[2][BM * LDA_S];
  __shared__ __align__(16) _Float16 sW[2][BN * LDW_S];
  const int bm = blockIdx.x * BM;
  const int bn = blockIdx.y * BN;
  const int t    = threadIdx.x;
  const int lane = t & 31;
  const int wid  = t >> 5;
  const int wm   = (wid >> 1) * 32;   // 4 waves in M
  const int wn   = (wid & 1) * 32;    // 2 waves in N
  const int l15  = lane & 15;
  const int hi   = lane >> 4;

  const int arow = t >> 3;            // 0..31
  const int acol = (t & 7) * 8;       // 16B chunks

  v8f acc00 = {}, acc01 = {}, acc10 = {}, acc11 = {};

  // 6 async b128 per thread per K-tile (4 for A, 2 for W)
  auto issue_tile = [&](int kt, int buf) {
    const _Float16* gA = A + (size_t)(bm + arow) * lda + kt + acol;
#pragma unroll
    for (int p = 0; p < 4; ++p)
      async_copy_b128(gA + (size_t)(32 * p) * lda, &sA[buf][(arow + 32 * p) * LDA_S + acol]);
    async_copy_b128(&W[(size_t)(bn + arow) * ldw + kt + acol],        &sW[buf][arow * LDA_S + acol]);
    async_copy_b128(&W[(size_t)(bn + arow + 32) * ldw + kt + acol],   &sW[buf][(arow + 32) * LDA_S + acol]);
  };

  const int ntiles = K / BK;
  issue_tile(0, 0);
  for (int ti = 0; ti < ntiles; ++ti) {
    const int buf = ti & 1;
    if (ti + 1 < ntiles) {
      issue_tile((ti + 1) * BK, buf ^ 1);
      asm volatile("s_wait_asynccnt 0x6" ::: "memory");  // tile ti landed; ti+1 in flight
    } else {
      asm volatile("s_wait_asynccnt 0x0" ::: "memory");
    }
    __syncthreads();
#pragma unroll
    for (int kk = 0; kk < BK; kk += 32) {
      v16h a0 = load_a_frag(sA[buf], wm,      kk, l15, hi);
      v16h a1 = load_a_frag(sA[buf], wm + 16, kk, l15, hi);
      v16h b0 = load_b_frag(sW[buf], wn,      kk, l15, hi);
      v16h b1 = load_b_frag(sW[buf], wn + 16, kk, l15, hi);
      acc00 = __builtin_amdgcn_wmma_f32_16x16x32_f16(false, a0, false, b0, (short)0, acc00, false, false);
      acc01 = __builtin_amdgcn_wmma_f32_16x16x32_f16(false, a0, false, b1, (short)0, acc01, false, false);
      acc10 = __builtin_amdgcn_wmma_f32_16x16x32_f16(false, a1, false, b0, (short)0, acc10, false, false);
      acc11 = __builtin_amdgcn_wmma_f32_16x16x32_f16(false, a1, false, b1, (short)0, acc11, false, false);
    }
    __syncthreads();  // all waves done with sA/sW[buf] before it is refilled
  }

  auto store_tile = [&](const v8f& acc, int ro, int co) {
    const int col = bn + wn + co + l15;
    const float bv = bias ? bias[col] : 0.f;
#pragma unroll
    for (int i = 0; i < 8; ++i) {
      const int row = bm + wm + ro + i + 8 * hi;   // C/D: VGPR i -> M = i + 8*hi
      C[(size_t)row * ldc + col] = (OutT)(acc[i] + bv);
    }
  };
  store_tile(acc00, 0, 0);
  store_tile(acc01, 0, 16);
  store_tile(acc10, 16, 0);
  store_tile(acc11, 16, 16);
}

// ---- elementwise / prep kernels ----
// 4 elements per thread (b128 load, b64 store)
__global__ void cvt_f16_kernel(const float* __restrict__ x, _Float16* __restrict__ y, size_t n4) {
  size_t i = (size_t)blockIdx.x * blockDim.x + threadIdx.x;
  if (i >= n4) return;
  float4 v = ((const float4*)x)[i];
  v4h h = {(_Float16)v.x, (_Float16)v.y, (_Float16)v.z, (_Float16)v.w};
  ((v4h*)y)[i] = h;
}

// out[k*D + j] = in[j*D + k]  (Wv -> Wv^T, f16)
__global__ void transpose_cvt_kernel(const float* __restrict__ in, _Float16* __restrict__ out) {
  size_t i = (size_t)blockIdx.x * blockDim.x + threadIdx.x;  // over D*D
  int j = (int)(i >> 10), k = (int)(i & 1023);
  out[(size_t)k * kDim + j] = (_Float16)in[i];
}

// cat16[m, :D] = f16(x1[m,:]);  cat16[m, D:] = f16(a[m,:] + b?[m,:]); 4 elems/thread
__global__ void pack_cat_kernel(const float* __restrict__ x1, const float* __restrict__ a,
                                const float* __restrict__ b, _Float16* __restrict__ cat) {
  size_t i = (size_t)blockIdx.x * blockDim.x + threadIdx.x;  // over B*D/4
  size_t e = i * 4;
  size_t m = e >> 10;
  int    c = (int)(e & 1023);
  float4 xv = ((const float4*)x1)[i];
  v4h xh = {(_Float16)xv.x, (_Float16)xv.y, (_Float16)xv.z, (_Float16)xv.w};
  *(v4h*)&cat[m * 2048 + c] = xh;
  float4 av = ((const float4*)a)[i];
  if (b) {
    float4 bv = ((const float4*)b)[i];
    av.x += bv.x; av.y += bv.y; av.z += bv.z; av.w += bv.w;
  }
  v4h rh = {(_Float16)av.x, (_Float16)av.y, (_Float16)av.z, (_Float16)av.w};
  *(v4h*)&cat[m * 2048 + 1024 + c] = rh;
}

// bc[i][n] = dot(w_out[i,n,:], b_in[i, 2D:3D]) + b_out[i,n]
__global__ __launch_bounds__(256) void bias_combine_kernel(const float* __restrict__ w_out,
    const float* __restrict__ b_in, const float* __restrict__ b_out, float* __restrict__ bc) {
  __shared__ float red[256];
  const int i = blockIdx.x >> 10;
  const int n = blockIdx.x & 1023;
  const float* wrow = w_out + ((size_t)i * kDim + n) * kDim;
  const float* bv   = b_in + (size_t)i * 3 * kDim + 2 * kDim;
  float s = 0.f;
  for (int c = threadIdx.x; c < kDim; c += 256) s += wrow[c] * bv[c];
  red[threadIdx.x] = s; __syncthreads();
  for (int o = 128; o > 0; o >>= 1) { if (threadIdx.x < o) red[threadIdx.x] += red[threadIdx.x + o]; __syncthreads(); }
  if (threadIdx.x == 0) bc[(size_t)i * kDim + n] = red[0] + b_out[(size_t)i * kDim + n];
}

// out = LN(base + sigmoid(gpre+gb) * (t1+t2[+t3])) ; one row per block
__global__ __launch_bounds__(256) void gate_ln_kernel(
    const float* __restrict__ base, const float* __restrict__ gpre,
    const float* __restrict__ gb,   const float* __restrict__ t1,
    const float* __restrict__ t2,   const float* __restrict__ t3,
    const float* __restrict__ lns,  const float* __restrict__ lnb,
    float* __restrict__ out32, _Float16* __restrict__ out16) {
  __shared__ float red[256];
  const int t = threadIdx.x;
  const size_t off = (size_t)blockIdx.x * kDim;
  float v[4]; float s = 0.f, ss = 0.f;
#pragma unroll
  for (int j = 0; j < 4; ++j) {
    const int c = t + j * 256;
    float g = gpre[off + c] + gb[c];
    g = 1.f / (1.f + __expf(-g));
    float sum = t1[off + c] + t2[off + c];
    if (t3) sum += t3[off + c];
    float val = base[off + c] + g * sum;
    v[j] = val; s += val; ss += val * val;
  }
  red[t] = s; __syncthreads();
  for (int o = 128; o > 0; o >>= 1) { if (t < o) red[t] += red[t + o]; __syncthreads(); }
  const float S = red[0]; __syncthreads();
  red[t] = ss; __syncthreads();
  for (int o = 128; o > 0; o >>= 1) { if (t < o) red[t] += red[t + o]; __syncthreads(); }
  const float SS  = red[0];
  const float mu  = S * (1.f / kDim);
  const float var = SS * (1.f / kDim) - mu * mu;
  const float rin = rsqrtf(var + kEps);
#pragma unroll
  for (int j = 0; j < 4; ++j) {
    const int c = t + j * 256;
    const float y = (v[j] - mu) * rin * lns[c] + lnb[c];
    out32[off + c] = y;
    if (out16) out16[off + c] = (_Float16)y;
  }
}

extern "C" void kernel_launch(void* const* d_in, const int* in_sizes, int n_in,
                              void* d_out, int out_size, void* d_ws, size_t ws_size,
                              hipStream_t stream) {
  (void)in_sizes; (void)n_in; (void)out_size; (void)ws_size;
  const float* text      = (const float*)d_in[0];
  const float* vision    = (const float*)d_in[1];
  const float* audio     = (const float*)d_in[2];
  const float* mha_w_in  = (const float*)d_in[3];
  const float* mha_b_in  = (const float*)d_in[4];
  const float* mha_w_out = (const float*)d_in[5];
  const float* mha_b_out = (const float*)d_in[6];
  const float* gate_w    = (const float*)d_in[7];
  const float* gate_b    = (const float*)d_in[8];
  const float* ln_scale  = (const float*)d_in[9];
  const float* ln_bias   = (const float*)d_in[10];
  float* out = (float*)d_out;

  const size_t BD = (size_t)kBatch * kDim;
  const size_t DD = (size_t)kDim * kDim;

  char* p = (char*)d_ws;
  auto alloc = [&](size_t bytes) { char* r = p; p += (bytes + 255) & ~(size_t)255; return r; };
  _Float16* text16   = (_Float16*)alloc(BD * 2);
  _Float16* vision16 = (_Float16*)alloc(BD * 2);
  _Float16* audio16  = (_Float16*)alloc(BD * 2);
  _Float16* te16     = (_Float16*)alloc(BD * 2);
  _Float16* cat16    = (_Float16*)alloc(BD * 2 * 2);     // [B, 2D]
  _Float16* Wc16     = (_Float16*)alloc(6 * DD * 2);
  _Float16* G16      = (_Float16*)alloc(3 * DD * 2 * 2); // [3, D, 2D]
  _Float16* WvT16    = (_Float16*)alloc(DD * 2);
  _Float16* Wout16   = (_Float16*)alloc(DD * 2);
  float* bc   = (float*)alloc(6 * kDim * 4);
  float* t2v  = (float*)alloc(BD * 4);   // reused as v2a in stage 3
  float* v2t  = (float*)alloc(BD * 4);
  float* t2a  = (float*)alloc(BD * 4);
  float* a2t  = (float*)alloc(BD * 4);
  float* a2v  = (float*)alloc(BD * 4);
  float* te   = (float*)alloc(BD * 4);
  float* gpre = (float*)alloc(BD * 4);

  const unsigned cvtBD4 = (unsigned)(BD / 4 / 256);
  const unsigned cvtDD  = (unsigned)(DD / 256);
  dim3 mgrid(kBatch / BM, kDim / BN);
  dim3 wgrid(kDim / BM, kDim / BN);
  auto Wc  = [&](int i) { return Wc16 + (size_t)i * DD; };
  auto Gw  = [&](int g) { return G16 + (size_t)g * 2 * DD; };

  // ---- prep: f16 activations & weights, combined Wc = W_out @ W_v, bc ----
  cvt_f16_kernel<<<cvtBD4, 256, 0, stream>>>(text,   text16,   BD / 4);
  cvt_f16_kernel<<<cvtBD4, 256, 0, stream>>>(vision, vision16, BD / 4);
  cvt_f16_kernel<<<cvtBD4, 256, 0, stream>>>(audio,  audio16,  BD / 4);
  cvt_f16_kernel<<<(unsigned)(6 * DD / 4 / 256), 256, 0, stream>>>(gate_w, G16, 6 * DD / 4);
  bias_combine_kernel<<<6 * kDim, 256, 0, stream>>>(mha_w_out, mha_b_in, mha_b_out, bc);
  for (int i = 0; i < 6; ++i) {
    const float* wv = mha_w_in + ((size_t)i * 3 * kDim + 2 * kDim) * kDim;
    transpose_cvt_kernel<<<cvtDD, 256, 0, stream>>>(wv, WvT16);
    cvt_f16_kernel<<<(unsigned)(DD / 4 / 256), 256, 0, stream>>>(mha_w_out + (size_t)i * DD, Wout16, DD / 4);
    // Wc[n,k] = sum_j Wout[n,j] * WvT[k,j]
    gemm_wmma<_Float16><<<wgrid, 256, 0, stream>>>(Wout16, kDim, WvT16, kDim, Wc(i), kDim, nullptr, kDim);
  }

  // ---- phase 1: GEMMs on raw inputs ----
  gemm_wmma<float><<<mgrid, 256, 0, stream>>>(vision16, kDim, Wc(0), kDim, t2v, kDim, bc + 0 * kDim, kDim);
  gemm_wmma<float><<<mgrid, 256, 0, stream>>>(text16,   kDim, Wc(1), kDim, v2t, kDim, bc + 1 * kDim, kDim);
  gemm_wmma<float><<<mgrid, 256, 0, stream>>>(audio16,  kDim, Wc(2), kDim, t2a, kDim, bc + 2 * kDim, kDim);
  gemm_wmma<float><<<mgrid, 256, 0, stream>>>(vision16, kDim, Wc(5), kDim, a2v, kDim, bc + 5 * kDim, kDim);

  // ---- stage 1: text_enhanced ----
  pack_cat_kernel<<<cvtBD4, 256, 0, stream>>>(text, v2t, nullptr, cat16);
  gemm_wmma<float><<<mgrid, 256, 0, stream>>>(cat16, 2 * kDim, Gw(0), 2 * kDim, gpre, kDim, nullptr, 2 * kDim);
  gate_ln_kernel<<<kBatch, 256, 0, stream>>>(text, gpre, gate_b, t2v, v2t, nullptr,
                                             ln_scale, ln_bias, te, te16);

  // ---- stage 2: text_final (t2v buffer becomes v2a after this point) ----
  gemm_wmma<float><<<mgrid, 256, 0, stream>>>(te16,    kDim, Wc(3), kDim, a2t, kDim, bc + 3 * kDim, kDim);
  gemm_wmma<float><<<mgrid, 256, 0, stream>>>(audio16, kDim, Wc(4), kDim, t2v, kDim, bc + 4 * kDim, kDim); // v2a
  pack_cat_kernel<<<cvtBD4, 256, 0, stream>>>(te, a2t, nullptr, cat16);
  gemm_wmma<float><<<mgrid, 256, 0, stream>>>(cat16, 2 * kDim, Gw(0), 2 * kDim, gpre, kDim, nullptr, 2 * kDim);
  gate_ln_kernel<<<kBatch, 256, 0, stream>>>(te, gpre, gate_b, t2a, a2t, nullptr,
                                             ln_scale, ln_bias, out, nullptr);

  // ---- stage 3: vision_final ----
  pack_cat_kernel<<<cvtBD4, 256, 0, stream>>>(vision, a2v, v2t, cat16);
  gemm_wmma<float><<<mgrid, 256, 0, stream>>>(cat16, 2 * kDim, Gw(1), 2 * kDim, gpre, kDim, nullptr, 2 * kDim);
  gate_ln_kernel<<<kBatch, 256, 0, stream>>>(vision, gpre, gate_b + kDim, t2v /*v2a*/, a2v, v2t,
                                             ln_scale + kDim, ln_bias + kDim, out + BD, nullptr);

  // ---- stage 3: audio_final ----
  pack_cat_kernel<<<cvtBD4, 256, 0, stream>>>(audio, a2t, a2v, cat16);
  gemm_wmma<float><<<mgrid, 256, 0, stream>>>(cat16, 2 * kDim, Gw(2), 2 * kDim, gpre, kDim, nullptr, 2 * kDim);
  gate_ln_kernel<<<kBatch, 256, 0, stream>>>(audio, gpre, gate_b + 2 * kDim, a2t, a2v, nullptr,
                                             ln_scale + 2 * kDim, ln_bias + 2 * kDim, out + 2 * BD, nullptr);
}
